// SRNN_16037407883648
// MI455X (gfx1250) — matrix operand order repaired and checked by reading, not compiled, not run on hardware
//
#include <hip/hip_runtime.h>
#include <hip/hip_bf16.h>
#include <stdint.h>

// ---------------------------------------------------------------------------
// SRNN on MI455X (gfx1250, wave32, WMMA).
//   Phase A (parallel): PRE[t] = xs_t @ Wi + sxs_t @ Ws  as grid-wide bf16
//                       WMMA GEMMs (f32 accumulate), PRE stored in WMMA
//                       C-fragment layout (contiguous 32B per lane).
//   Phase B (serial)  : h = tanh(PRE[t] + h @ Wh) in ONE 32-wave workgroup,
//                       h in LDS (bf16, 64 KB, XOR bank swizzle), 2048 WMMAs
//                       per step, s_barrier per step, Wh fragments prefetched.
// Weights pre-swizzled into WMMA B-fragment order (two b128 loads per lane).
// ---------------------------------------------------------------------------

typedef __attribute__((ext_vector_type(16))) __bf16        v16bf;
typedef __attribute__((ext_vector_type(8)))  float         v8f;
typedef __attribute__((ext_vector_type(4)))  unsigned int  u32x4;
typedef __attribute__((ext_vector_type(4)))  float         f32x4;

union Frag { v16bf v; u32x4 q[2]; };
union AccU { v8f  v; f32x4 q[2]; };

#define T_   512
#define B_   64
#define F_   128
#define H_   512
#define NT_  (H_/16)   // 32 column tiles

__device__ __forceinline__ unsigned short f2bf(float f) {
    unsigned int u = __float_as_uint(f);
    u += 0x7FFFu + ((u >> 16) & 1u);        // round-to-nearest-even
    return (unsigned short)(u >> 16);
}

// ---------------------------------------------------------------------------
// Build bf16 time-major input xT(t,b,f) and moving averages sx2 / sx4.
// src x is (B,T,F) f32.
// ---------------------------------------------------------------------------
__global__ void srnn_prep_x(const float* __restrict__ x,
                            unsigned short* __restrict__ xT,
                            unsigned short* __restrict__ sx2,
                            unsigned short* __restrict__ sx4) {
    long idx = (long)blockIdx.x * 256 + threadIdx.x;     // over T*B*F
    if (idx >= (long)T_ * B_ * F_) return;
    int f = (int)(idx & (F_ - 1));
    long r = idx >> 7;                                   // F_=128
    int b = (int)(r & (B_ - 1));
    int t = (int)(r >> 6);                               // B_=64
    const float* p = x + ((long)b * T_ + t) * F_ + f;
    float v0 = p[0];
    xT[idx] = f2bf(v0);
    if (t < T_ - 1) {
        float v1 = p[F_];
        sx2[idx] = f2bf((v0 + v1) * 0.5f);
        if (t < T_ - 3) {
            float v2 = p[2 * F_], v3 = p[3 * F_];
            sx4[idx] = f2bf((v0 + v1 + v2 + v3) * 0.25f);
        }
    }
}

// ---------------------------------------------------------------------------
// Swizzle an f32 KxN weight matrix into WMMA B-fragment-major bf16 packing.
// Fragment (kt,nt) = 32x16 tile; per lane 8 dwords contiguous (32B).
// ISA B layout: lanes 0-15 hold K=kt*32+0..7,16..23 (pairs per dword), lanes 16-31
// hold K=kt*32+8..15,24..31; column N = nt*16 + (lane&15).
// ---------------------------------------------------------------------------
__global__ void srnn_pack_w(const float* __restrict__ W,
                            unsigned int* __restrict__ dst, int K, int N) {
    long w = (long)blockIdx.x * 256 + threadIdx.x;
    if (w >= (long)K * N / 2) return;
    int  j    = (int)(w & 7);
    int  lane = (int)((w >> 3) & 31);
    long frag = w >> 8;
    int  NT   = N / 16;
    int  nt   = (int)(frag % NT);
    int  kt   = (int)(frag / NT);
    int  n    = nt * 16 + (lane & 15);
    int  k    = kt * 32 + ((lane >> 4) << 3) + (j < 4 ? 2 * j : 16 + 2 * (j - 4));
    unsigned int lo = f2bf(W[(long)k * N + n]);
    unsigned int hi = f2bf(W[(long)(k + 1) * N + n]);
    dst[w] = lo | (hi << 16);
}

// ---------------------------------------------------------------------------
// PRE tile (t, bt, nt) stored fragment-major: base + (((t*4+bt)*32+nt)*32
// + lane)*8 floats, i.e. 32B contiguous per lane -> two b128 per access.
// ---------------------------------------------------------------------------
__device__ __forceinline__ long pre_off(long t, int bt, int nt, int lane) {
    return ((((t * 4 + bt) * 32) + nt) * 32 + lane) * 8;
}

// ---------------------------------------------------------------------------
// PRE[t] = XS[t] @ Wi  +  SXS[t] @ Ws     (grid-parallel, bf16 WMMA, f32 acc)
// XS: (n,B,Kin) bf16 row-major (shift-adjusted), SXS: (n,B,128) bf16.
// Each wave computes one 16x16 f32 tile of PRE.
// ---------------------------------------------------------------------------
__global__ void __launch_bounds__(256)
srnn_pre_gemm(const unsigned short* __restrict__ XS, int Kin,
              const unsigned short* __restrict__ SXS,
              const unsigned int* __restrict__ WIp,
              const unsigned int* __restrict__ WSp,
              float* __restrict__ PREf, int n) {
    int  lane = threadIdx.x & 31;
    int  wave = threadIdx.x >> 5;
    long tile = (long)blockIdx.x * 8 + wave;
    int  nt   = (int)(tile & 31);
    int  bt   = (int)((tile >> 5) & 3);
    long t    = tile >> 7;
    if (t >= n) return;

    int mrow = bt * 16 + (lane & 15);
    int kb   = (lane >> 4) * 8;          // A-fragment K sub-offset (0 or 8)
    AccU acc; acc.v = (v8f){};

    // ---- XS @ Wi -----------------------------------------------------------
    const unsigned short* arow = XS + ((t * B_ + mrow) * (long)Kin);
    for (int kt = 0; kt < Kin / 32; ++kt) {
        Frag A, Bm;
        A.q[0] = *(const u32x4*)(arow + kt * 32 + kb);
        A.q[1] = *(const u32x4*)(arow + kt * 32 + kb + 16);
        const unsigned int* bp = WIp + (((long)kt * NT_ + nt) * 32 + lane) * 8;
        Bm.q[0] = *(const u32x4*)(bp);
        Bm.q[1] = *(const u32x4*)(bp + 4);
        acc.v = __builtin_amdgcn_wmma_f32_16x16x32_bf16(false, A.v, false, Bm.v,
                                                        (short)0, acc.v, false, false);
    }
    // ---- SXS @ Ws (Ks = 128) ----------------------------------------------
    const unsigned short* srow = SXS + ((t * B_ + mrow) * (long)F_);
    for (int kt = 0; kt < F_ / 32; ++kt) {
        Frag A, Bm;
        A.q[0] = *(const u32x4*)(srow + kt * 32 + kb);
        A.q[1] = *(const u32x4*)(srow + kt * 32 + kb + 16);
        const unsigned int* bp = WSp + (((long)kt * NT_ + nt) * 32 + lane) * 8;
        Bm.q[0] = *(const u32x4*)(bp);
        Bm.q[1] = *(const u32x4*)(bp + 4);
        acc.v = __builtin_amdgcn_wmma_f32_16x16x32_bf16(false, A.v, false, Bm.v,
                                                        (short)0, acc.v, false, false);
    }
    // ---- store in fragment layout: 32B contiguous per lane -----------------
    float* op = PREf + pre_off(t, bt, nt, lane);
    *(f32x4*)(op)     = acc.q[0];
    *(f32x4*)(op + 4) = acc.q[1];
}

// ---------------------------------------------------------------------------
// Serial recurrence: h = tanh(PRE[t] + h @ Wh), t = 0..n-1. One workgroup,
// 32 waves. h in LDS as bf16 with 16B-chunk XOR swizzle (chunk ^= m&15).
// Wave w owns m-tile (w&3) and n-tiles {(w>>2)+8i}. Writes f32 to OUT and a
// bf16 copy to CUR (next layer's GEMM operand).
// ---------------------------------------------------------------------------
__device__ __forceinline__ const unsigned short*
hswz(const unsigned short* base, int m, int chunk) {
    return base + m * H_ + ((chunk ^ (m & 15)) << 3);
}

__global__ void __launch_bounds__(1024)
srnn_recur(const float* __restrict__ PREf,
           const unsigned int* __restrict__ WHp,
           float* __restrict__ OUT,
           unsigned short* __restrict__ CUR, int n) {
    __shared__ unsigned short hbuf[B_ * H_];             // 64 KB

    int tid  = threadIdx.x;
    int lane = tid & 31;
    int wave = tid >> 5;
    for (int i = tid; i < B_ * H_; i += 1024) hbuf[i] = 0;   // h0 = 0
    __syncthreads();

    int mtile = wave & 3;
    int ngrp  = wave >> 2;                               // 0..7
    int m     = mtile * 16 + (lane & 15);                // A-fragment row
    int hi    = lane >> 4;
    int coll  = lane & 15;

    for (int t = 0; t < n; ++t) {
        AccU acc[4];
        // init accumulators from PRE (fragment layout: two b128 per tile)
#pragma unroll
        for (int i = 0; i < 4; ++i) {
            const float* p = PREf + pre_off(t, mtile, ngrp + 8 * i, lane);
            acc[i].q[0] = *(const f32x4*)(p);
            acc[i].q[1] = *(const f32x4*)(p + 4);
        }
        // h @ Wh : K = 512 -> 16 k-tiles, A from LDS, B from packed global
        for (int kt = 0; kt < 16; ++kt) {
            // prefetch next k-tile's first B fragment (global_prefetch_b8)
            if (kt < 15) {
                const unsigned int* pf =
                    WHp + (((long)(kt + 1) * NT_ + ngrp) * 32 + lane) * 8;
                __builtin_prefetch(pf, 0, 3);
            }
            Frag A;
            int c0 = kt * 4 + hi;                        // 16B chunk index
            A.q[0] = *(const u32x4*)hswz(hbuf, m, c0);
            A.q[1] = *(const u32x4*)hswz(hbuf, m, c0 + 2);
#pragma unroll
            for (int i = 0; i < 4; ++i) {
                int nt = ngrp + 8 * i;
                const unsigned int* bp = WHp + (((long)kt * NT_ + nt) * 32 + lane) * 8;
                Frag Bm;
                Bm.q[0] = *(const u32x4*)(bp);
                Bm.q[1] = *(const u32x4*)(bp + 4);
                acc[i].v = __builtin_amdgcn_wmma_f32_16x16x32_bf16(
                    false, A.v, false, Bm.v, (short)0, acc[i].v, false, false);
            }
        }
        // tanh, emit f32 output + bf16 copy, stash new h
        unsigned short hb[4][8];
#pragma unroll
        for (int i = 0; i < 4; ++i) {
            int nt = ngrp + 8 * i;
            float*          op = OUT + ((long)t * B_ + mtile * 16) * H_ + nt * 16 + coll;
            unsigned short* cp = CUR + ((long)t * B_ + mtile * 16) * H_ + nt * 16 + coll;
#pragma unroll
            for (int j = 0; j < 8; ++j) {
                float v = tanhf(acc[i].v[j]);
                op[(j + 8 * hi) * H_] = v;
                unsigned short b = f2bf(v);
                hb[i][j] = b;
                cp[(j + 8 * hi) * H_] = b;
            }
        }
        __syncthreads();                                  // old h fully consumed
#pragma unroll
        for (int i = 0; i < 4; ++i) {
            int nt = ngrp + 8 * i;
#pragma unroll
            for (int j = 0; j < 8; ++j) {
                int mr = mtile * 16 + j + 8 * hi;
                int nc = nt * 16 + coll;
                int ch = nc >> 3;
                hbuf[mr * H_ + (((ch ^ (mr & 15)) << 3) | (nc & 7))] = hb[i][j];
            }
        }
        __syncthreads();                                  // new h visible
    }
}

// ---------------------------------------------------------------------------
extern "C" void kernel_launch(void* const* d_in, const int* in_sizes, int n_in,
                              void* d_out, int out_size, void* d_ws, size_t ws_size,
                              hipStream_t stream) {
    (void)in_sizes; (void)n_in; (void)out_size; (void)ws_size;
    const float* x = (const float*)d_in[0];

    // ---- workspace carve-up ------------------------------------------------
    size_t off = 0;
    auto carve = [&](size_t bytes) {
        void* p = (char*)d_ws + off;
        off = (off + bytes + 255) & ~(size_t)255;
        return p;
    };
    unsigned short* xT  = (unsigned short*)carve((size_t)T_ * B_ * F_ * 2);
    unsigned short* sx2 = (unsigned short*)carve((size_t)(T_-1) * B_ * F_ * 2);
    unsigned short* sx4 = (unsigned short*)carve((size_t)(T_-3) * B_ * F_ * 2);

    // packed weights: order wi0,ws0,wh0,wi1,ws1,wh1,wi2,ws2,wh2 (d_in[1..9])
    const int Kdim[9] = {128,128,512, 512,128,512, 512,128,512};
    unsigned int* Wp[9];
    for (int i = 0; i < 9; ++i)
        Wp[i] = (unsigned int*)carve((size_t)Kdim[i] * H_ / 2 * 4);

    float*          PREf = (float*)carve((size_t)(T_-1) * B_ * H_ * 4);
    unsigned short* CUR  = (unsigned short*)carve((size_t)(T_-1) * B_ * H_ * 2);

    // ---- phase 0: input prep + weight swizzle ------------------------------
    srnn_prep_x<<<(T_*B_*F_ + 255)/256, 256, 0, stream>>>(x, xT, sx2, sx4);
    for (int i = 0; i < 9; ++i) {
        long words = (long)Kdim[i] * H_ / 2;
        srnn_pack_w<<<(int)((words + 255)/256), 256, 0, stream>>>(
            (const float*)d_in[1 + i], Wp[i], Kdim[i], H_);
    }

    const int n0 = 511, n1 = 509, n2 = 505;
    float* out0 = (float*)d_out;
    float* out1 = out0 + (size_t)n0 * B_ * H_;
    float* out2 = out1 + (size_t)n1 * B_ * H_;

    // ---- layer 0 (scale 1): xs = x[0:511], sxs = x[0:511] ------------------
    srnn_pre_gemm<<<(n0*128 + 7)/8, 256, 0, stream>>>(xT, 128, xT,
                                                      Wp[0], Wp[1], PREf, n0);
    srnn_recur<<<1, 1024, 0, stream>>>(PREf, Wp[2], out0, CUR, n0);

    // ---- layer 1 (scale 2): xs = out0[1:510], sxs = sx2[1:510] -------------
    srnn_pre_gemm<<<(n1*128 + 7)/8, 256, 0, stream>>>(
        CUR + (size_t)1 * B_ * H_, 512, sx2 + (size_t)1 * B_ * F_,
        Wp[3], Wp[4], PREf, n1);
    srnn_recur<<<1, 1024, 0, stream>>>(PREf, Wp[5], out1, CUR, n1);

    // ---- layer 2 (scale 4): xs = out1[3:508], sxs = sx4[3:508] -------------
    srnn_pre_gemm<<<(n2*128 + 7)/8, 256, 0, stream>>>(
        CUR + (size_t)3 * B_ * H_, 512, sx4 + (size_t)3 * B_ * F_,
        Wp[6], Wp[7], PREf, n2);
    srnn_recur<<<1, 1024, 0, stream>>>(PREf, Wp[8], out2, CUR, n2);
}
